// NeuralODE_59253368816104
// MI455X (gfx1250) — compile-verified, matrix operand order checked by reading, not run on hardware
//
#include <hip/hip_runtime.h>

typedef __attribute__((ext_vector_type(16))) _Float16 v16h;
typedef __attribute__((ext_vector_type(8)))  _Float16 v8h;
typedef __attribute__((ext_vector_type(8)))  float    v8f;

#define N_STEPS 1024
#define HSTR    520   // LDS row stride (halfs) for 16x512 activation tiles (bank-spread)
#define XSTR    72    // LDS row stride (halfs) for 16x64 input tile

// f16 weight workspace layout (element offsets into d_ws)
#define W0P_OFF 0
#define W1P_OFF (512*64)
#define W2P_OFF (W1P_OFF + 512*512)
#define W3P_OFF (W2P_OFF + 512*512)

// ---------------------------------------------------------------------------
// One-time (per launch) weight conversion: f32 -> f16, W0 zero-padded 40->64 K.
// Weights stored row-major [n][k] (i.e. already transposed for h @ W^T).
// ---------------------------------------------------------------------------
__global__ __launch_bounds__(256) void prep_weights_f16(
    const float* __restrict__ W0, const float* __restrict__ W1,
    const float* __restrict__ W2, const float* __restrict__ W3,
    _Float16* __restrict__ ws)
{
  int idx = blockIdx.x * 256 + threadIdx.x;
  if (idx < 512 * 64) {
    int n = idx >> 6, kc = idx & 63;
    ws[W0P_OFF + idx] = (_Float16)(kc < 40 ? W0[n * 40 + kc] : 0.0f);
  }
  if (idx < 512 * 512) {
    ws[W1P_OFF + idx] = (_Float16)W1[idx];
    ws[W2P_OFF + idx] = (_Float16)W2[idx];
  }
  if (idx < 32 * 512) {
    ws[W3P_OFF + idx] = (_Float16)W3[idx];
  }
}

// ---------------------------------------------------------------------------
// WMMA helpers (CDNA5 16x16x32 f16 -> f32, wave32)
// ---------------------------------------------------------------------------
__device__ __forceinline__ v8f wmma_f16(v16h a, v16h b, v8f c) {
  // 8 args: (neg_a, A, neg_b, B, c_mod, C, reuse_a, reuse_b)
  return __builtin_amdgcn_wmma_f32_16x16x32_f16(false, a, false, b, (short)0, c,
                                                false, false);
}

__device__ __forceinline__ v8f splat8(float v) {
  v8f c = {v, v, v, v, v, v, v, v};
  return c;
}

// leaky_relu(0.01) == max(x, 0.01x) for slope<1. Direct v_max_num_f32 via
// inline asm to avoid clang's extra NaN-canonicalize max(x,x).
__device__ __forceinline__ float leaky01(float x) {
  float r;
  float y = 0.01f * x;
  asm("v_max_num_f32 %0, %1, %2" : "=v"(r) : "v"(x), "v"(y));
  return r;
}

// A operand: 16(M) x 32(K) f16 tile from LDS, rows at `stride` halfs.
// Per ISA 7.12.2: lanes 0-15 hold M=lane, K = [kk..kk+7] ++ [kk+16..kk+23];
// lanes 16-31 hold M=lane-16, K offset +8. Two contiguous 16B LDS loads.
__device__ __forceinline__ v16h load_a_lds(const _Float16* buf, int stride,
                                           int kk, int lane) {
  int m   = lane & 15;
  int off = (lane >> 4) << 3;  // 0 or 8
  const _Float16* p = buf + m * stride + kk + off;
  v8h lo = *(const v8h*)(p);
  v8h hi = *(const v8h*)(p + 16);
  return __builtin_shufflevector(lo, hi, 0, 1, 2, 3, 4, 5, 6, 7,
                                 8, 9, 10, 11, 12, 13, 14, 15);
}

// B operand: 32(K) x 16(N) f16 tile. Weight stored row-major [n][k] so each
// lane (column n = n0 + lane%16) reads two contiguous 16B chunks of row n.
__device__ __forceinline__ v16h load_b_glb(const _Float16* __restrict__ W,
                                           int kdim, int n0, int kk, int lane) {
  int n   = n0 + (lane & 15);
  int off = (lane >> 4) << 3;
  const _Float16* p = W + n * kdim + kk + off;
  v8h lo = *(const v8h*)(p);
  v8h hi = *(const v8h*)(p + 16);
  return __builtin_shufflevector(lo, hi, 0, 1, 2, 3, 4, 5, 6, 7,
                                 8, 9, 10, 11, 12, 13, 14, 15);
}

// Epilogue: leaky + f16 store. C layout: VGPR j holds (m = j + 8*(lane>=16),
// n = lane&15). Bias is already folded into the accumulator init.
__device__ __forceinline__ void store_act(_Float16* dst, int stride, int n0,
                                          v8f acc, int lane) {
  int n = n0 + (lane & 15);
  int mbase = (lane >> 4) << 3;
#pragma unroll
  for (int j = 0; j < 8; ++j) {
    dst[(mbase + j) * stride + n] = (_Float16)leaky01(acc[j]);
  }
}

// ---------------------------------------------------------------------------
// Main rollout: one workgroup = 16 batch rows, 8 waves split the 512 hidden
// columns (64 each). Iterates all N_STEPS locally; weights stream from L2
// (total f16 weights ~1.1MB << 192MB L2; HBM only sees the 134MB trajectory).
// ---------------------------------------------------------------------------
__global__ __launch_bounds__(256) void ode_rollout(
    const float* __restrict__ init_obs, const float* __restrict__ actions,
    const _Float16* __restrict__ ws,
    const float* __restrict__ b0, const float* __restrict__ b1,
    const float* __restrict__ b2, const float* __restrict__ b3,
    const int* __restrict__ tau_p, float* __restrict__ out)
{
  __shared__ _Float16 xbuf[16 * XSTR];   // 16 x 64 padded input (y|a|0)
  __shared__ _Float16 hA[16 * HSTR];     // activation ping
  __shared__ _Float16 hB[16 * HSTR];     // activation pong

  const _Float16* __restrict__ W0p = ws + W0P_OFF;
  const _Float16* __restrict__ W1p = ws + W1P_OFF;
  const _Float16* __restrict__ W2p = ws + W2P_OFF;
  const _Float16* __restrict__ W3p = ws + W3P_OFF;

  const int tid   = threadIdx.x;
  const int lane  = tid & 31;
  const int wave  = tid >> 5;
  const int brow0 = blockIdx.x * 16;
  const float tauf = (float)tau_p[0];

  const int ncol  = (lane & 15);          // n within a 16-wide tile
  const int mbase = (lane >> 4) << 3;     // 0 or 8 (C-tile row base)
  const int nw    = wave * 64;            // this wave's hidden-column base

  // --- loop-invariant biases -> registers (accumulator init values) ---
  float b0r[4], b1r[4], b2r[4];
#pragma unroll
  for (int t = 0; t < 4; ++t) {
    b0r[t] = b0[nw + t * 16 + ncol];
    b1r[t] = b1[nw + t * 16 + ncol];
    b2r[t] = b2[nw + t * 16 + ncol];
  }

  // --- state y lives in registers of waves 0-1 (C-tile layout) ---
  float yreg[8];
  float b3r = 0.0f;
  if (wave < 2) {
    int n = wave * 16 + ncol;             // obs column 0..31
    b3r = b3[n];
#pragma unroll
    for (int j = 0; j < 8; ++j) {
      int m = mbase + j;
      float v = init_obs[(brow0 + m) * 32 + n];
      yreg[j] = v;
      out[((size_t)(brow0 + m) * (N_STEPS + 1)) * 32 + n] = v;       // traj[:,0,:]
      xbuf[m * XSTR + n] = (_Float16)v;                              // x y-part, k=0
    }
  }
  // zero pad cols 40..63 (never overwritten afterwards; disjoint addresses)
  for (int i = tid; i < 16 * 24; i += 256) {
    int m = i / 24, c = 40 + (i % 24);
    xbuf[m * XSTR + c] = (_Float16)0.0f;
  }
  // actions for k=0 (cols 32..39)
  if (tid < 128) {
    int m = tid >> 3, j = tid & 7;
    xbuf[m * XSTR + 32 + j] = (_Float16)actions[((size_t)(brow0 + m) * N_STEPS) * 8 + j];
  }

  for (int k = 0; k < N_STEPS; ++k) {
    __syncthreads();  // xbuf (y | a | pad) for step k is complete

    // ---- layer0: x(16x64) @ W0p^T -> hA (16x512), each wave 64 cols ----
    {
      v8f a0 = splat8(b0r[0]), a1 = splat8(b0r[1]);
      v8f a2 = splat8(b0r[2]), a3 = splat8(b0r[3]);
#pragma unroll
      for (int kk = 0; kk < 64; kk += 32) {
        v16h av = load_a_lds(xbuf, XSTR, kk, lane);
        a0 = wmma_f16(av, load_b_glb(W0p, 64, nw +  0, kk, lane), a0);
        a1 = wmma_f16(av, load_b_glb(W0p, 64, nw + 16, kk, lane), a1);
        a2 = wmma_f16(av, load_b_glb(W0p, 64, nw + 32, kk, lane), a2);
        a3 = wmma_f16(av, load_b_glb(W0p, 64, nw + 48, kk, lane), a3);
      }
      store_act(hA, HSTR, nw +  0, a0, lane);
      store_act(hA, HSTR, nw + 16, a1, lane);
      store_act(hA, HSTR, nw + 32, a2, lane);
      store_act(hA, HSTR, nw + 48, a3, lane);
    }
    __syncthreads();

    // ---- layer1: hA @ W1p^T -> hB ----
    {
      v8f a0 = splat8(b1r[0]), a1 = splat8(b1r[1]);
      v8f a2 = splat8(b1r[2]), a3 = splat8(b1r[3]);
#pragma unroll 4
      for (int kk = 0; kk < 512; kk += 32) {
        v16h av = load_a_lds(hA, HSTR, kk, lane);
        a0 = wmma_f16(av, load_b_glb(W1p, 512, nw +  0, kk, lane), a0);
        a1 = wmma_f16(av, load_b_glb(W1p, 512, nw + 16, kk, lane), a1);
        a2 = wmma_f16(av, load_b_glb(W1p, 512, nw + 32, kk, lane), a2);
        a3 = wmma_f16(av, load_b_glb(W1p, 512, nw + 48, kk, lane), a3);
      }
      store_act(hB, HSTR, nw +  0, a0, lane);
      store_act(hB, HSTR, nw + 16, a1, lane);
      store_act(hB, HSTR, nw + 32, a2, lane);
      store_act(hB, HSTR, nw + 48, a3, lane);
    }
    __syncthreads();

    // ---- layer2: hB @ W2p^T -> hA ----
    {
      v8f a0 = splat8(b2r[0]), a1 = splat8(b2r[1]);
      v8f a2 = splat8(b2r[2]), a3 = splat8(b2r[3]);
#pragma unroll 4
      for (int kk = 0; kk < 512; kk += 32) {
        v16h av = load_a_lds(hB, HSTR, kk, lane);
        a0 = wmma_f16(av, load_b_glb(W2p, 512, nw +  0, kk, lane), a0);
        a1 = wmma_f16(av, load_b_glb(W2p, 512, nw + 16, kk, lane), a1);
        a2 = wmma_f16(av, load_b_glb(W2p, 512, nw + 32, kk, lane), a2);
        a3 = wmma_f16(av, load_b_glb(W2p, 512, nw + 48, kk, lane), a3);
      }
      store_act(hA, HSTR, nw +  0, a0, lane);
      store_act(hA, HSTR, nw + 16, a1, lane);
      store_act(hA, HSTR, nw + 32, a2, lane);
      store_act(hA, HSTR, nw + 48, a3, lane);
    }
    __syncthreads();

    // ---- layer3 + Euler (waves 0-1); waves 2-5 prefetch actions[k+1] ----
    if (wave < 2) {
      const int n0 = wave * 16;
      v8f acc = splat8(b3r);
#pragma unroll 4
      for (int kk = 0; kk < 512; kk += 32) {
        v16h av = load_a_lds(hA, HSTR, kk, lane);
        acc = wmma_f16(av, load_b_glb(W3p, 512, n0, kk, lane), acc);
      }
      int n = n0 + ncol;
#pragma unroll
      for (int j = 0; j < 8; ++j) {
        int m = mbase + j;
        float ynew = fmaf(tauf, acc[j], yreg[j]);
        yreg[j] = ynew;
        out[((size_t)(brow0 + m) * (N_STEPS + 1) + (k + 1)) * 32 + n] = ynew;
        xbuf[m * XSTR + n] = (_Float16)ynew;   // x y-part for step k+1
      }
    } else if (wave < 6) {
      // 128 lanes load next step's actions into xbuf cols 32..39
      int idx = (wave - 2) * 32 + lane;        // 0..127
      int m = idx >> 3, j = idx & 7;
      int ka = (k + 1 < N_STEPS) ? (k + 1) : (N_STEPS - 1);
      xbuf[m * XSTR + 32 + j] =
          (_Float16)actions[((size_t)(brow0 + m) * N_STEPS + ka) * 8 + j];
    }
  }
}

// ---------------------------------------------------------------------------
extern "C" void kernel_launch(void* const* d_in, const int* in_sizes, int n_in,
                              void* d_out, int out_size, void* d_ws, size_t ws_size,
                              hipStream_t stream) {
  const float* init_obs = (const float*)d_in[0];
  const float* actions  = (const float*)d_in[1];
  const float* W0 = (const float*)d_in[2];
  const float* b0 = (const float*)d_in[3];
  const float* W1 = (const float*)d_in[4];
  const float* b1 = (const float*)d_in[5];
  const float* W2 = (const float*)d_in[6];
  const float* b2 = (const float*)d_in[7];
  const float* W3 = (const float*)d_in[8];
  const float* b3 = (const float*)d_in[9];
  const int*  tau = (const int*)d_in[10];
  float* out = (float*)d_out;
  _Float16* ws = (_Float16*)d_ws;

  // Convert weights to f16 tiles in workspace (re-done every call; deterministic).
  hipLaunchKernelGGL(prep_weights_f16, dim3((512 * 512 + 255) / 256), dim3(256),
                     0, stream, W0, W1, W2, W3, ws);

  // 64 workgroups x 256 threads (8 wave32s): each WG owns 16 batch rows.
  hipLaunchKernelGGL(ode_rollout, dim3(1024 / 16), dim3(256), 0, stream,
                     init_obs, actions, ws, b0, b1, b2, b3, tau, out);
}
